// CompoundDecoder_90598040142400
// MI455X (gfx1250) — compile-verified, matrix operand order checked by reading, not run on hardware
//
#include <hip/hip_runtime.h>

// ---------------------------------------------------------------------------
// CompoundDecoder: persistent per-batch-tile LSTM decode on gfx1250 (CDNA5).
// 2048 batch rows, 3 LSTM layers (H=256), 99 autoregressive steps.
// M=32 rows per workgroup (64 WGs x 512 threads = 16 waves): each wave owns a
// 16-column gate slice (all 4 gates) and BOTH 16-row M-tiles, so every B
// (weight) fragment fetched from L2 feeds two v_wmma_f32_16x16x32_bf16 ops.
// c-state lives in VGPRs; h ping-pongs through dynamic LDS as bf16.
// Gate nonlinearities use v_exp_f32/v_rcp_f32 (TRANS, co-executes with WMMA);
// argmax is a half-wave shuffle butterfly instead of a serial scan.
// ---------------------------------------------------------------------------

typedef __attribute__((ext_vector_type(16))) __bf16 v16bf;
typedef __attribute__((ext_vector_type(8)))  float  v8f;

#define HID   256
#define INDIM 512
#define XCATW 768          // HID + INDIM (layer-0 input width)
#define VOCAB 100
#define VPAD  112          // vocab padded to 7 WMMA tiles
#define MAXL  100
#define BATCH 2048
#define MT    32           // batch rows per workgroup
#define NBLK  (BATCH / MT) // 64 workgroups
#define NTHR  512          // 16 waves

// dynamic LDS layout (bytes)
#define SM_XCAT   0                                   // [MT][XCATW] bf16 = 49152
#define SM_HBUF   (MT * XCATW * 2)                    // [2][3][MT][HID] bf16 = 98304
#define SM_LOGIT  (SM_HBUF + 2 * 3 * MT * HID * 2)    // [MT][VPAD] f32 = 14336
#define SM_TOTAL  (SM_LOGIT + MT * VPAD * 4)

__device__ __forceinline__ unsigned short f2bf(float x) {
    unsigned int u = __float_as_uint(x);
    unsigned int r = u + 0x7fffu + ((u >> 16) & 1u);   // round-to-nearest-even
    return (unsigned short)(r >> 16);
}

__device__ __forceinline__ v8f vzero8() {
    v8f z = {0.f,0.f,0.f,0.f,0.f,0.f,0.f,0.f};
    return z;
}

union FragU { uint4 u4[2]; v16bf v; };

// A fragment (16x32 bf16, M x K) from LDS. Lane m = l&15; lane-half selects K
// blocks {0..7,16..23} vs {8..15,24..31}. Caller pre-offsets buf by M-tile.
__device__ __forceinline__ v16bf load_a_lds(const unsigned short* buf, int stride,
                                            int lane, int kb) {
    int m  = lane & 15;
    int kh = (lane >> 4) << 3;               // 0 or 8
    const unsigned short* p = buf + m * stride + kb + kh;
    FragU f;
    f.u4[0] = *(const uint4*)(p);
    f.u4[1] = *(const uint4*)(p + 16);
    return f.v;
}

// B fragment (32x16 bf16, K x N): weights row-major W[n][k]; lane n = l&15.
__device__ __forceinline__ v16bf load_b_glb(const unsigned short* w, int stride,
                                            int lane, int nrow_base, int kb) {
    int n  = lane & 15;
    int kh = (lane >> 4) << 3;
    const unsigned short* p = w + (size_t)(nrow_base + n) * stride + kb + kh;
    FragU f;
    f.u4[0] = *(const uint4*)(p);
    f.u4[1] = *(const uint4*)(p + 16);
    return f.v;
}

#define WMMA_BF16(a, b, c) \
    __builtin_amdgcn_wmma_f32_16x16x32_bf16(false, (a), false, (b), (short)0, (c), false, false)

// Fast transcendental gates: v_exp_f32 + v_rcp_f32 (both TRANS ops).
__device__ __forceinline__ float fast_sigmoid(float x) {
    return __builtin_amdgcn_rcpf(1.0f + __expf(-x));
}
__device__ __forceinline__ float fast_tanh(float x) {
    // tanh(x) = 2*sigmoid(2x) - 1 ; saturates correctly at +-1
    return __builtin_amdgcn_rcpf(1.0f + __expf(-2.0f * x)) * 2.0f - 1.0f;
}

// ---------------------------------------------------------------------------
// Prep kernels: fp32 -> bf16 conversion + bias fusion + vocab padding
// ---------------------------------------------------------------------------
__global__ void cvt_bf16_kernel(const float* __restrict__ s,
                                unsigned short* __restrict__ d, int n) {
    int i = blockIdx.x * 256 + threadIdx.x;
    if (i < n) d[i] = f2bf(s[i]);
}

__global__ void prep_wout_kernel(const float* __restrict__ w_out,
                                 const float* __restrict__ b_out,
                                 unsigned short* __restrict__ wd,
                                 float* __restrict__ bd) {
    int i = blockIdx.x * 256 + threadIdx.x;
    if (i < VPAD * HID) {
        int r = i / HID;
        wd[i] = (r < VOCAB) ? f2bf(w_out[i]) : (unsigned short)0;
    }
    if (i < VPAD) bd[i] = (i < VOCAB) ? b_out[i] : -1e30f;
}

__global__ void prep_bias_kernel(const float* b0i, const float* b0h,
                                 const float* b1i, const float* b1h,
                                 const float* b2i, const float* b2h,
                                 float* __restrict__ bc) {
    int i = blockIdx.x * 256 + threadIdx.x;
    if (i >= 3 * 4 * HID) return;
    int l = i >> 10, k = i & 1023;
    const float* bi = (l == 0) ? b0i : (l == 1) ? b1i : b2i;
    const float* bh = (l == 0) ? b0h : (l == 1) ? b1h : b2h;
    bc[i] = bi[k] + bh[k];
}

// ---------------------------------------------------------------------------
// Main persistent decode kernel: grid = 64 blocks x 512 threads (16 waves).
// Wave w owns hidden columns [16w, 16w+16) across all 4 gates, both M-tiles.
// ---------------------------------------------------------------------------
__global__ __launch_bounds__(NTHR) void decode_kernel(
    const float* __restrict__ z,               // [2048][512]
    const unsigned short* __restrict__ embb,   // [100][256] bf16
    const float* __restrict__ b_lat,           // [256]
    const unsigned short* __restrict__ wlatb,  // [256][512] bf16
    const unsigned short* __restrict__ wih0,   // [1024][768] bf16
    const unsigned short* __restrict__ whh0,   // [1024][256] bf16
    const unsigned short* __restrict__ wih1,   // [1024][256] bf16
    const unsigned short* __restrict__ whh1,   // [1024][256] bf16
    const unsigned short* __restrict__ wih2,   // [1024][256] bf16
    const unsigned short* __restrict__ whh2,   // [1024][256] bf16
    const unsigned short* __restrict__ woutb,  // [112][256] bf16 (padded)
    const float* __restrict__ bcomb,           // [3][1024] f32 (b_ih + b_hh)
    const float* __restrict__ boutp,           // [112] f32 (-1e30 pad)
    int* __restrict__ out)                     // tokens [2048][100] + pads[2048]
{
    extern __shared__ __align__(16) char smem[];
    unsigned short* xcat   = (unsigned short*)(smem + SM_XCAT);   // [MT][XCATW]
    unsigned short* hbufB  = (unsigned short*)(smem + SM_HBUF);   // [2][3][MT][HID]
    float*          logits = (float*)(smem + SM_LOGIT);           // [MT][VPAD]
    __shared__ int wtok[MT], eosf[MT], padsA[MT];

    const int tid  = threadIdx.x;
    const int lane = tid & 31;
    const int wv   = tid >> 5;          // 0..15
    const int b0   = blockIdx.x * MT;
    const int ncol = 16 * wv + (lane & 15);   // this lane's gate/hidden column
    const int mofs = (lane >> 4) << 3;        // intra-tile row offset (0 or 8)

    auto hptr = [&](int p, int l) -> unsigned short* {
        return hbufB + ((p * 3 + l) * MT * HID);
    };

    // ---- step-invariant per-lane biases (hoisted out of the 99-step loop) --
    float blat_r = b_lat[ncol];
    float bg_r[3][4];
    #pragma unroll
    for (int l = 0; l < 3; ++l)
        #pragma unroll
        for (int gi = 0; gi < 4; ++gi)
            bg_r[l][gi] = bcomb[l * 1024 + gi * HID + ncol];
    float bo_r = (wv < 7) ? boutp[16 * wv + (lane & 15)] : 0.f;

    // ---- one-time: stage z rows into xcat (bf16) --------------------------
    for (int e = tid; e < MT * INDIM; e += NTHR) {
        int m = e >> 9, k = e & (INDIM - 1);
        xcat[m * XCATW + HID + k] = f2bf(z[(size_t)(b0 + m) * INDIM + k]);
    }
    if (tid < MT) {
        wtok[tid] = 1; eosf[tid] = 0; padsA[tid] = MAXL;
        out[(b0 + tid) * MAXL + 0] = 1;                 // x_full[:,0] = 1
    }
    __syncthreads();

    // ---- h0 = z @ w_lat^T + b_lat ; h[l]=c[l]=h0 (write parity 0) ---------
    v8f cst[3][2];                                      // [layer][M-tile] c-state
    {
        v8f acc0 = vzero8(), acc1 = vzero8();
        for (int kb = 0; kb < INDIM; kb += 32) {
            v16bf bb = load_b_glb(wlatb, INDIM, lane, 16 * wv, kb);
            v16bf a0 = load_a_lds(xcat + HID, XCATW, lane, kb);
            v16bf a1 = load_a_lds(xcat + HID + 16 * XCATW, XCATW, lane, kb);
            acc0 = WMMA_BF16(a0, bb, acc0);
            acc1 = WMMA_BF16(a1, bb, acc1);
        }
        #pragma unroll
        for (int mt = 0; mt < 2; ++mt) {
            const v8f& a = mt ? acc1 : acc0;
            #pragma unroll
            for (int j = 0; j < 8; ++j) {
                float h0v = a[j] + blat_r;
                cst[0][mt][j] = h0v; cst[1][mt][j] = h0v; cst[2][mt][j] = h0v;
                unsigned short hv = f2bf(h0v);
                int m = mt * 16 + mofs + j;
                hptr(0, 0)[m * HID + ncol] = hv;
                hptr(0, 1)[m * HID + ncol] = hv;
                hptr(0, 2)[m * HID + ncol] = hv;
            }
        }
    }
    __syncthreads();

    // ---- 99 autoregressive steps ------------------------------------------
    for (int step = 1; step < MAXL; ++step) {
        const int p = step & 1;                  // write parity this step

        // embedding of previous token -> xcat[:, 0:256) (bf16, 32B per thread)
        {
            int m  = tid >> 4;                   // 0..31
            int c0 = (tid & 15) << 4;            // 0..240
            const uint4* er = (const uint4*)(embb + (size_t)wtok[m] * HID + c0);
            uint4* xw = (uint4*)(xcat + m * XCATW + c0);
            xw[0] = er[0];
            xw[1] = er[1];
        }
        __syncthreads();

        for (int l = 0; l < 3; ++l) {
            const unsigned short* xnew = (l == 0) ? xcat : hptr(p, l - 1);
            const unsigned short* hold = hptr(p ^ 1, l);
            const int xstride = (l == 0) ? XCATW : HID;
            const int Kih     = (l == 0) ? XCATW : HID;
            const unsigned short* wih = (l == 0) ? wih0 : (l == 1) ? wih1 : wih2;
            const unsigned short* whh = (l == 0) ? whh0 : (l == 1) ? whh1 : whh2;

            v8f acc[4][2];                       // [gate][M-tile]
            #pragma unroll
            for (int gi = 0; gi < 4; ++gi) { acc[gi][0] = vzero8(); acc[gi][1] = vzero8(); }

            // g += x @ W_ih^T   (B fragment shared by both M-tiles)
            for (int kb = 0; kb < Kih; kb += 32) {
                v16bf a0 = load_a_lds(xnew, xstride, lane, kb);
                v16bf a1 = load_a_lds(xnew + 16 * xstride, xstride, lane, kb);
                #pragma unroll
                for (int gi = 0; gi < 4; ++gi) {
                    v16bf bb = load_b_glb(wih, Kih, lane, gi * HID + 16 * wv, kb);
                    acc[gi][0] = WMMA_BF16(a0, bb, acc[gi][0]);
                    acc[gi][1] = WMMA_BF16(a1, bb, acc[gi][1]);
                }
            }
            // g += h_prev @ W_hh^T
            for (int kb = 0; kb < HID; kb += 32) {
                v16bf a0 = load_a_lds(hold, HID, lane, kb);
                v16bf a1 = load_a_lds(hold + 16 * HID, HID, lane, kb);
                #pragma unroll
                for (int gi = 0; gi < 4; ++gi) {
                    v16bf bb = load_b_glb(whh, HID, lane, gi * HID + 16 * wv, kb);
                    acc[gi][0] = WMMA_BF16(a0, bb, acc[gi][0]);
                    acc[gi][1] = WMMA_BF16(a1, bb, acc[gi][1]);
                }
            }

            // gate fusion in registers; c never leaves VGPRs. Writes go to the
            // opposite parity buffer, so no barrier needed before writing.
            unsigned short* hw = hptr(p, l);
            #pragma unroll
            for (int mt = 0; mt < 2; ++mt) {
                #pragma unroll
                for (int j = 0; j < 8; ++j) {
                    float iv = fast_sigmoid(acc[0][mt][j] + bg_r[l][0]);
                    float fv = fast_sigmoid(acc[1][mt][j] + bg_r[l][1]);
                    float gv = fast_tanh  (acc[2][mt][j] + bg_r[l][2]);
                    float ov = fast_sigmoid(acc[3][mt][j] + bg_r[l][3]);
                    float cn = fv * cst[l][mt][j] + iv * gv;
                    cst[l][mt][j] = cn;
                    hw[(mt * 16 + mofs + j) * HID + ncol] = f2bf(ov * fast_tanh(cn));
                }
            }
            __syncthreads();   // h[p][l] complete before layer l+1 reads it
        }

        // logits = h2 @ w_out^T + b_out (7 WMMA column tiles over waves 0..6)
        if (wv < 7) {
            const unsigned short* h2 = hptr(p, 2);
            v8f acc0 = vzero8(), acc1 = vzero8();
            for (int kb = 0; kb < HID; kb += 32) {
                v16bf bb = load_b_glb(woutb, HID, lane, 16 * wv, kb);
                v16bf a0 = load_a_lds(h2, HID, lane, kb);
                v16bf a1 = load_a_lds(h2 + 16 * HID, HID, lane, kb);
                acc0 = WMMA_BF16(a0, bb, acc0);
                acc1 = WMMA_BF16(a1, bb, acc1);
            }
            int vcol = 16 * wv + (lane & 15);
            #pragma unroll
            for (int mt = 0; mt < 2; ++mt) {
                const v8f& a = mt ? acc1 : acc0;
                #pragma unroll
                for (int j = 0; j < 8; ++j)
                    logits[(mt * 16 + mofs + j) * VPAD + vcol] = a[j] + bo_r;
            }
        }
        __syncthreads();

        // parallel greedy argmax: 16 lanes per row scan 7 strided elements,
        // then a 4-step shuffle butterfly within each half-wave.
        // Tie rule matches jnp.argmax: greater wins; equal -> lower index.
        {
            int m = tid >> 4;                    // row handled by this half-wave
            int s = tid & 15;
            float best = -3e38f;
            int   bidx = VPAD;
            #pragma unroll
            for (int k = 0; k < 7; ++k) {
                int v2 = s + (k << 4);
                float x = logits[m * VPAD + v2];
                if (x > best) { best = x; bidx = v2; }
            }
            #pragma unroll
            for (int d = 8; d >= 1; d >>= 1) {
                float ov = __shfl_xor(best, d, 16);
                int   oi = __shfl_xor(bidx, d, 16);
                if (ov > best || (ov == best && oi < bidx)) { best = ov; bidx = oi; }
            }
            if (s == 0) {
                int e = eosf[m];
                out[(b0 + m) * MAXL + step] = e ? 0 : bidx;
                if (!e && bidx == 2) { eosf[m] = 1; padsA[m] = step + 1; }
                wtok[m] = bidx;
            }
        }
        __syncthreads();
    }

    if (tid < MT) out[BATCH * MAXL + b0 + tid] = padsA[tid];
}

// ---------------------------------------------------------------------------
extern "C" void kernel_launch(void* const* d_in, const int* in_sizes, int n_in,
                              void* d_out, int out_size, void* d_ws, size_t ws_size,
                              hipStream_t stream) {
    const float* z     = (const float*)d_in[0];
    const float* emb   = (const float*)d_in[1];
    const float* w_lat = (const float*)d_in[2];
    const float* b_lat = (const float*)d_in[3];
    const float* w_out = (const float*)d_in[4];
    const float* b_out = (const float*)d_in[5];
    const float* wih_f[3], *whh_f[3], *bih_f[3], *bhh_f[3];
    for (int l = 0; l < 3; ++l) {
        wih_f[l] = (const float*)d_in[6 + 4 * l + 0];
        whh_f[l] = (const float*)d_in[6 + 4 * l + 1];
        bih_f[l] = (const float*)d_in[6 + 4 * l + 2];
        bhh_f[l] = (const float*)d_in[6 + 4 * l + 3];
    }

    // workspace layout (bf16 weights + fused biases)
    char* w = (char*)d_ws;
    size_t off = 0;
    auto take = [&](size_t bytes) { char* p = w + off; off += (bytes + 255) & ~size_t(255); return p; };
    unsigned short* wlatb = (unsigned short*)take(HID * INDIM * 2);          // 256x512
    unsigned short* wih0b = (unsigned short*)take(4 * HID * XCATW * 2);      // 1024x768
    unsigned short* whh0b = (unsigned short*)take(4 * HID * HID * 2);        // 1024x256
    unsigned short* wih1b = (unsigned short*)take(4 * HID * HID * 2);
    unsigned short* whh1b = (unsigned short*)take(4 * HID * HID * 2);
    unsigned short* wih2b = (unsigned short*)take(4 * HID * HID * 2);
    unsigned short* whh2b = (unsigned short*)take(4 * HID * HID * 2);
    unsigned short* woutb = (unsigned short*)take(VPAD * HID * 2);           // 112x256
    unsigned short* embb  = (unsigned short*)take(VOCAB * HID * 2);          // 100x256
    float*          bcomb = (float*)take(3 * 4 * HID * 4);                   // 3x1024
    float*          boutp = (float*)take(VPAD * 4);
    (void)ws_size; (void)in_sizes; (void)n_in; (void)out_size;

    auto cvt = [&](const float* s, unsigned short* d, int n) {
        cvt_bf16_kernel<<<(n + 255) / 256, 256, 0, stream>>>(s, d, n);
    };
    cvt(w_lat,    wlatb, HID * INDIM);
    cvt(wih_f[0], wih0b, 4 * HID * XCATW);
    cvt(whh_f[0], whh0b, 4 * HID * HID);
    cvt(wih_f[1], wih1b, 4 * HID * HID);
    cvt(whh_f[1], whh1b, 4 * HID * HID);
    cvt(wih_f[2], wih2b, 4 * HID * HID);
    cvt(whh_f[2], whh2b, 4 * HID * HID);
    cvt(emb,      embb,  VOCAB * HID);
    prep_wout_kernel<<<(VPAD * HID + 255) / 256, 256, 0, stream>>>(w_out, b_out, woutb, boutp);
    prep_bias_kernel<<<(3 * 4 * HID + 255) / 256, 256, 0, stream>>>(
        bih_f[0], bhh_f[0], bih_f[1], bhh_f[1], bih_f[2], bhh_f[2], bcomb);

    // dynamic LDS exceeds the 64KB default; CDNA5 allows 320KB per workgroup
    (void)hipFuncSetAttribute((const void*)decode_kernel,
                              hipFuncAttributeMaxDynamicSharedMemorySize, SM_TOTAL);
    decode_kernel<<<NBLK, NTHR, SM_TOTAL, stream>>>(
        z, embb, b_lat, wlatb,
        wih0b, whh0b, wih1b, whh1b, wih2b, whh2b,
        woutb, bcomb, boutp, (int*)d_out);
}
